// AsymmetricAttention_7481833029784
// MI455X (gfx1250) — matrix operand order repaired and checked by reading, not compiled, not run on hardware
//
#include <hip/hip_runtime.h>
#include <hip/hip_fp16.h>

typedef _Float16 half_t;
typedef __attribute__((ext_vector_type(16))) _Float16 v16h;
typedef __attribute__((ext_vector_type(8)))  _Float16 v8h;
typedef __attribute__((ext_vector_type(8)))  float    v8f;
typedef __attribute__((ext_vector_type(4)))  unsigned int u32x4;
typedef __attribute__((ext_vector_type(8)))  int i32x8;
typedef __attribute__((ext_vector_type(4)))  int i32x4;

// ---------------- problem constants ----------------
constexpr int Nx = 2048;
constexpr int Ly = 256;
constexpr int H  = 24;
constexpr int D  = 128;
constexpr int DX = H * D;            // 3072
constexpr int DY = 1536;
constexpr int LV = 200;
constexpr int TOTAL = Nx + LV;       // 2248 valid attention tokens
constexpr int AP = 2272;             // padded to multiple of 32
constexpr int OUT_ROWS = Nx + Ly;    // 2304 rows fed to projections
constexpr float EPS = 1e-6f;
constexpr float SM_SCALE = 0.08838834764831845f;  // 1/sqrt(128)

// ---------------- WMMA wrapper ----------------
__device__ __forceinline__ v8f wmma_f16(v16h a, v16h b, v8f c) {
  // (neg_a, A, neg_b, B, c_mod, C, reuse_a, reuse_b)
  return __builtin_amdgcn_wmma_f32_16x16x32_f16(false, a, false, b, (short)0, c,
                                                false, false);
}

__device__ __forceinline__ float warpSum32(float v) {
#pragma unroll
  for (int o = 16; o > 0; o >>= 1) v += __shfl_xor(v, o, 32);
  return v;
}

// ---------------- TDM: 2D tile global->LDS DMA ----------------
// Loads a tile of 32 halfs x `rows` rows (row stride K halfs in memory) into
// LDS at byte offset lds_addr, inserting 4 DWORDs of padding after every
// 16 DWORDs (i.e. after each 32-half row) -> LDS row stride = 40 halfs.
__device__ __forceinline__ void tdm_load_tile_2d(unsigned int lds_addr,
                                                 const half_t* gptr,
                                                 unsigned int K,
                                                 unsigned int rows) {
  u32x4 g0;
  g0[0] = 1u;  // count=1 (valid user descriptor), no gather
  g0[1] = lds_addr;
  const unsigned long long ga = (unsigned long long)gptr;
  g0[2] = (unsigned int)ga;
  g0[3] = (unsigned int)((ga >> 32) & 0x01FFFFFFull) | (2u << 30);  // type=2

  i32x8 g1;
  // data_size=1 (2B), pad_enable=1, pad_interval=3 (16 DW), pad_amount=3 (4 DW)
  g1[0] = (int)((1u << 16) | (1u << 20) | (3u << 22) | (3u << 25));
  g1[1] = (int)((K & 0xFFFFu) << 16);                              // dim0 lo
  g1[2] = (int)(((K >> 16) & 0xFFFFu) | ((rows & 0xFFFFu) << 16)); // dim0 hi|dim1 lo
  g1[3] = (int)(((rows >> 16) & 0xFFFFu) | (32u << 16));           // dim1 hi|tile_dim0=32
  g1[4] = (int)(rows & 0xFFFFu);                                   // tile_dim1=rows
  g1[5] = (int)K;                                                  // dim0_stride lo
  g1[6] = 0;
  g1[7] = 0;

  i32x4 z4;
  z4[0] = z4[1] = z4[2] = z4[3] = 0;
#if __clang_major__ >= 23
  i32x8 z8;
#pragma unroll
  for (int i = 0; i < 8; ++i) z8[i] = 0;
  __builtin_amdgcn_tensor_load_to_lds(g0, g1, z4, z4, z8, 0);
#else
  __builtin_amdgcn_tensor_load_to_lds(g0, g1, z4, z4, 0);
#endif
}

// ---------------- kernel 1: rmsnorm * (1+scale) -> f16 ----------------
__global__ __launch_bounds__(256) void rmsnorm_scale_cvt(
    const float* __restrict__ X, const float* __restrict__ scale,
    half_t* __restrict__ out, int Dims) {
  const int row = blockIdx.x;
  const float* xr = X + (size_t)row * Dims;
  float ss = 0.f;
  for (int c = threadIdx.x; c < Dims; c += 256) {
    float v = xr[c];
    ss += v * v;
  }
  __shared__ float s8[8];
  ss = warpSum32(ss);
  if ((threadIdx.x & 31) == 0) s8[threadIdx.x >> 5] = ss;
  __syncthreads();
  float tot = 0.f;
#pragma unroll
  for (int i = 0; i < 8; ++i) tot += s8[i];
  const float r = rsqrtf(tot / (float)Dims + EPS);
  for (int c = threadIdx.x; c < Dims; c += 256)
    out[(size_t)row * Dims + c] = (half_t)(xr[c] * r * (1.f + scale[c]));
}

// ---------------- kernel 2: f32 -> f16 convert / zero fill ----------------
__global__ __launch_bounds__(256) void cvt_f32_f16(const float* __restrict__ in,
                                                   half_t* __restrict__ out,
                                                   size_t n) {
  size_t i = (size_t)blockIdx.x * 256 + threadIdx.x;
  size_t stride = (size_t)gridDim.x * 256;
  for (; i < n; i += stride) out[i] = (half_t)in[i];
}

__global__ __launch_bounds__(256) void fill_zero_f16(half_t* __restrict__ p,
                                                     size_t n) {
  size_t i = (size_t)blockIdx.x * 256 + threadIdx.x;
  if (i < n) p[i] = (half_t)0.f;
}

// ---------------- kernel 3: WMMA GEMM  C = A * W^T + bias ----------------
// A: [M][K] f16 row-major, W: [Nt][K] f16 row-major, C: [M][Nt] f32
// Block tile 128x128x32, 256 threads = 8 waves (2x4), each wave 64x32 output.
// Tiles staged global->LDS by the Tensor Data Mover, double-buffered.
constexpr int LDSS = 40;  // LDS half-stride: 32 data + 4 DW TDM padding

__global__ __launch_bounds__(256) void gemm_f16_wmma(
    const half_t* __restrict__ A, const half_t* __restrict__ W,
    const float* __restrict__ bias, float* __restrict__ C, int M, int Nt,
    int K) {
  __shared__ half_t sA[2][128 * LDSS];
  __shared__ half_t sB[2][128 * LDSS];
  const int tid = threadIdx.x;
  const int lane = tid & 31;
  const int wave = tid >> 5;
  const int m0 = blockIdx.y * 128;
  const int n0 = blockIdx.x * 128;
  const int wm = (wave >> 2) * 64;  // 0 / 64
  const int wn = (wave & 3) * 32;   // 0..96
  const int hl = lane & 15;
  const int hi8 = (lane >> 4) * 8;
  const int kh16 = (lane >> 4) * 16;

  const unsigned int ldsA0 = (unsigned int)(unsigned long long)&sA[0][0];
  const unsigned int ldsA1 = (unsigned int)(unsigned long long)&sA[1][0];
  const unsigned int ldsB0 = (unsigned int)(unsigned long long)&sB[0][0];
  const unsigned int ldsB1 = (unsigned int)(unsigned long long)&sB[1][0];

  v8f acc[4][2];
#pragma unroll
  for (int i = 0; i < 4; ++i)
#pragma unroll
    for (int j = 0; j < 2; ++j)
#pragma unroll
      for (int r = 0; r < 8; ++r) acc[i][j][r] = 0.f;

  const int nk = K / 32;
  const half_t* gA = A + (size_t)m0 * K;
  const half_t* gB = W + (size_t)n0 * K;

  if (wave == 0) {  // TDM issue is wave-level (EXEC ignored); one wave issues
    tdm_load_tile_2d(ldsA0, gA, (unsigned)K, 128u);
    tdm_load_tile_2d(ldsB0, gB, (unsigned)K, 128u);
  }

  for (int i = 0; i < nk; ++i) {
    const int buf = i & 1;
    if (wave == 0) {
      if (i + 1 < nk) {
        // prefetch next tile into the other buffer, then wait for current
        tdm_load_tile_2d(buf ? ldsA0 : ldsA1, gA + (size_t)(i + 1) * 32,
                         (unsigned)K, 128u);
        tdm_load_tile_2d(buf ? ldsB0 : ldsB1, gB + (size_t)(i + 1) * 32,
                         (unsigned)K, 128u);
        __builtin_amdgcn_s_wait_tensorcnt(2);
      } else {
        __builtin_amdgcn_s_wait_tensorcnt(0);
      }
    }
    __syncthreads();  // current buffer is ready for all waves

    const half_t* cA = &sA[buf][0];
    const half_t* cB = &sB[buf][0];
    // A fragments: lane m holds row m, K in {hi8+0..7} U {16+hi8+0..7}
    v16h af[4];
#pragma unroll
    for (int t = 0; t < 4; ++t) {
      const int row = wm + t * 16 + hl;
      const v8h p0 = *(const v8h*)&cA[row * LDSS + hi8];
      const v8h p1 = *(const v8h*)&cA[row * LDSS + 16 + hi8];
#pragma unroll
      for (int e = 0; e < 8; ++e) {
        af[t][e] = p0[e];
        af[t][8 + e] = p1[e];
      }
    }
    // B fragments: lane n holds W-row (output col) n, K in {hi8+0..7} U {16+hi8+0..7}
    v16h bf[2];
#pragma unroll
    for (int j = 0; j < 2; ++j) {
      const int row = wn + j * 16 + hl;
      const v8h p0 = *(const v8h*)&cB[row * LDSS + hi8];
      const v8h p1 = *(const v8h*)&cB[row * LDSS + 16 + hi8];
#pragma unroll
      for (int e = 0; e < 8; ++e) {
        bf[j][e] = p0[e];
        bf[j][8 + e] = p1[e];
      }
    }
#pragma unroll
    for (int t = 0; t < 4; ++t)
#pragma unroll
      for (int j = 0; j < 2; ++j) acc[t][j] = wmma_f16(af[t], bf[j], acc[t][j]);
    __syncthreads();  // all waves done reading before buffer is overwritten
  }

  // epilogue: C/D layout -> global f32 + bias
  const int nn = lane & 15;
  const int mb = (lane >> 4) * 8;
#pragma unroll
  for (int t = 0; t < 4; ++t)
#pragma unroll
    for (int j = 0; j < 2; ++j) {
      const int col = n0 + wn + j * 16 + nn;
      const float b = bias[col];
#pragma unroll
      for (int r = 0; r < 8; ++r) {
        const int row = m0 + wm + t * 16 + mb + r;
        C[(size_t)row * Nt + col] = acc[t][j][r] + b;
      }
    }
}

// ---------------- kernel 4: QKV post-process ----------------
// per-head rmsnorm + q/k norm weight + RoPE (x stream) + gather + pad.
// q is pre-scaled by 1/sqrt(D). v is stored transposed [H][D][AP].
__global__ __launch_bounds__(128) void qkv_prep(
    const float* __restrict__ qkvx, const float* __restrict__ qkvy,
    const float* __restrict__ qnx, const float* __restrict__ knx,
    const float* __restrict__ qny, const float* __restrict__ kny,
    const float* __restrict__ rcos, const float* __restrict__ rsin,
    const int* __restrict__ vti, half_t* __restrict__ qh,
    half_t* __restrict__ kh, half_t* __restrict__ vT) {
  const int t = blockIdx.x;   // 0..AP-1
  const int h = blockIdx.y;   // 0..H-1
  const int d = threadIdx.x;  // 0..127
  const size_t qkoff = ((size_t)t * H + h) * D + d;
  const size_t vToff = ((size_t)h * D + d) * AP + t;

  if (t >= TOTAL) {  // zero padding rows
    qh[qkoff] = (half_t)0.f;
    kh[qkoff] = (half_t)0.f;
    vT[vToff] = (half_t)0.f;
    return;
  }
  const int idx = vti[t];
  const bool is_x = idx < Nx;
  const float* src =
      is_x ? (qkvx + (size_t)idx * 3 * DX) : (qkvy + (size_t)(idx - Nx) * 3 * DX);
  const int hd = h * D + d;
  float qv = src[hd];
  float kv = src[DX + hd];
  float vv = src[2 * DX + hd];

  __shared__ float sq4[4], sk4[4];
  float wq = warpSum32(qv * qv);
  float wk = warpSum32(kv * kv);
  if ((threadIdx.x & 31) == 0) {
    sq4[threadIdx.x >> 5] = wq;
    sk4[threadIdx.x >> 5] = wk;
  }
  __syncthreads();
  const float ssq = sq4[0] + sq4[1] + sq4[2] + sq4[3];
  const float ssk = sk4[0] + sk4[1] + sk4[2] + sk4[3];
  const float qr = rsqrtf(ssq / (float)D + EPS);
  const float kr = rsqrtf(ssk / (float)D + EPS);
  float qn = qv * qr * (is_x ? qnx[d] : qny[d]);
  float kn = kv * kr * (is_x ? knx[d] : kny[d]);

  if (is_x) {  // RoPE on x stream, position idx
    const int i = d >> 1;
    const size_t ro = ((size_t)idx * H + h) * (D / 2) + i;
    const float c = rcos[ro];
    const float s = rsin[ro];
    const float oq = __shfl_xor(qn, 1, 32);
    const float ok = __shfl_xor(kn, 1, 32);
    qn = (d & 1) ? (oq * s + qn * c) : (qn * c - oq * s);
    kn = (d & 1) ? (ok * s + kn * c) : (kn * c - ok * s);
  }
  qh[qkoff] = (half_t)(qn * SM_SCALE);
  kh[qkoff] = (half_t)kn;
  vT[vToff] = (half_t)vv;
}

// ---------------- kernel 5: flash attention (WMMA, online softmax) --------
// 1 wave handles 16 q rows of one head. KV tiles of 32 tokens.
__global__ __launch_bounds__(32) void flash_attn(
    const half_t* __restrict__ q, const half_t* __restrict__ k,
    const half_t* __restrict__ vT, half_t* __restrict__ out) {
  __shared__ half_t sp[16 * LDSS];  // P staging tile (C-layout -> A-layout)
  const int lane = threadIdx.x;
  const int qt = blockIdx.x;  // q tile (16 rows)
  const int h = blockIdx.y;
  const int hl = lane & 15;
  const int hi8 = (lane >> 4) * 8;
  const int kh16 = (lane >> 4) * 16;
  const int mb = (lane >> 4) * 8;

  // q fragments (pre-scaled by 1/sqrt(D)): 4 chunks over D=128
  v16h qa[4];
#pragma unroll
  for (int c = 0; c < 4; ++c) {
    const half_t* p = q + ((size_t)(qt * 16 + hl) * H + h) * D + c * 32;
    const v8h p0 = *(const v8h*)(p + hi8);
    const v8h p1 = *(const v8h*)(p + 16 + hi8);
#pragma unroll
    for (int e = 0; e < 8; ++e) {
      qa[c][e] = p0[e];
      qa[c][8 + e] = p1[e];
    }
  }

  v8f O[8];
#pragma unroll
  for (int nc = 0; nc < 8; ++nc)
#pragma unroll
    for (int r = 0; r < 8; ++r) O[nc][r] = 0.f;
  float mr[8], lsum[8];
#pragma unroll
  for (int r = 0; r < 8; ++r) {
    mr[r] = -1e30f;
    lsum[r] = 0.f;
  }

  for (int kt = 0; kt < AP / 32; ++kt) {
    // S(16x32) = q(16x128) @ k_tile^T : two C fragments, 4 WMMAs each
    v8f S0, S1;
#pragma unroll
    for (int r = 0; r < 8; ++r) {
      S0[r] = 0.f;
      S1[r] = 0.f;
    }
#pragma unroll
    for (int j = 0; j < 2; ++j) {
#pragma unroll
      for (int c = 0; c < 4; ++c) {
        v16h kb;
        const half_t* p =
            k + ((size_t)(kt * 32 + j * 16 + hl) * H + h) * D + c * 32 + hi8;
        const v8h p0 = *(const v8h*)p;
        const v8h p1 = *(const v8h*)(p + 16);
#pragma unroll
        for (int e = 0; e < 8; ++e) {
          kb[e] = p0[e];
          kb[8 + e] = p1[e];
        }
        if (j == 0)
          S0 = wmma_f16(qa[c], kb, S0);
        else
          S1 = wmma_f16(qa[c], kb, S1);
      }
    }
    // mask padded key tokens (column token id per lane)
    const int tok0 = kt * 32 + hl;
    if (tok0 >= TOTAL)
#pragma unroll
      for (int r = 0; r < 8; ++r) S0[r] = -1e30f;
    if (tok0 + 16 >= TOTAL)
#pragma unroll
      for (int r = 0; r < 8; ++r) S1[r] = -1e30f;

    // online softmax, row stats via half-wave shuffles (rows live per 16 lanes)
    v8f P0, P1;
#pragma unroll
    for (int r = 0; r < 8; ++r) {
      float t = fmaxf(S0[r], S1[r]);
#pragma unroll
      for (int o = 8; o > 0; o >>= 1) t = fmaxf(t, __shfl_xor(t, o, 16));
      const float mn = fmaxf(mr[r], t);
      const float al = __expf(mr[r] - mn);
      const float p0 = __expf(S0[r] - mn);
      const float p1 = __expf(S1[r] - mn);
      float rs = p0 + p1;
#pragma unroll
      for (int o = 8; o > 0; o >>= 1) rs += __shfl_xor(rs, o, 16);
      lsum[r] = lsum[r] * al + rs;
      mr[r] = mn;
      P0[r] = p0;
      P1[r] = p1;
#pragma unroll
      for (int nc = 0; nc < 8; ++nc) O[nc][r] *= al;
    }

    // P: C-layout -> A-layout via LDS (DS ops are in-order within a wave)
#pragma unroll
    for (int r = 0; r < 8; ++r) {
      sp[(mb + r) * LDSS + hl] = (half_t)P0[r];
      sp[(mb + r) * LDSS + 16 + hl] = (half_t)P1[r];
    }
    v16h pa;
    {
      const v8h p0 = *(const v8h*)&sp[hl * LDSS + hi8];
      const v8h p1 = *(const v8h*)&sp[hl * LDSS + 16 + hi8];
#pragma unroll
      for (int e = 0; e < 8; ++e) {
        pa[e] = p0[e];
        pa[8 + e] = p1[e];
      }
    }
    // O(16x128) += P(16x32) @ V_tile(32x128); vT gives contiguous B columns
#pragma unroll
    for (int nc = 0; nc < 8; ++nc) {
      v16h vb;
      const half_t* p = vT + ((size_t)h * D + nc * 16 + hl) * AP + kt * 32 + hi8;
      const v8h p0 = *(const v8h*)p;
      const v8h p1 = *(const v8h*)(p + 16);
#pragma unroll
      for (int e = 0; e < 8; ++e) {
        vb[e] = p0[e];
        vb[8 + e] = p1[e];
      }
      O[nc] = wmma_f16(pa, vb, O[nc]);
    }
  }

  // epilogue: normalize and store f16 rows (skip padded q rows)
#pragma unroll
  for (int r = 0; r < 8; ++r) {
    const int row = qt * 16 + mb + r;
    if (row < TOTAL) {
      const float inv = 1.f / lsum[r];
#pragma unroll
      for (int nc = 0; nc < 8; ++nc)
        out[(size_t)row * DX + h * D + nc * 16 + hl] = (half_t)(O[nc][r] * inv);
    }
  }
}

// ---------------- host launcher ----------------
extern "C" void kernel_launch(void* const* d_in, const int* in_sizes, int n_in,
                              void* d_out, int out_size, void* d_ws,
                              size_t ws_size, hipStream_t stream) {
  (void)in_sizes;
  (void)n_in;
  (void)out_size;
  (void)ws_size;
  const float* x = (const float*)d_in[0];
  const float* y = (const float*)d_in[1];
  const float* scale_x = (const float*)d_in[2];
  const float* scale_y = (const float*)d_in[3];
  const float* rope_cos = (const float*)d_in[4];
  const float* rope_sin = (const float*)d_in[5];
  const float* W_qkv_x = (const float*)d_in[6];
  const float* b_qkv_x = (const float*)d_in[7];
  const float* W_qkv_y = (const float*)d_in[8];
  const float* b_qkv_y = (const float*)d_in[9];
  const float* q_norm_x = (const float*)d_in[10];
  const float* k_norm_x = (const float*)d_in[11];
  const float* q_norm_y = (const float*)d_in[12];
  const float* k_norm_y = (const float*)d_in[13];
  const float* W_proj_x = (const float*)d_in[14];
  const float* b_proj_x = (const float*)d_in[15];
  const float* W_proj_y = (const float*)d_in[16];
  const float* b_proj_y = (const float*)d_in[17];
  const int* vti = (const int*)d_in[18];
  float* out = (float*)d_out;

  // workspace carve-up (256B aligned)
  char* cur = (char*)d_ws;
  auto alloc = [&](size_t bytes) {
    void* p = (void*)cur;
    cur += (bytes + 255) & ~(size_t)255;
    return p;
  };
  half_t* xm = (half_t*)alloc((size_t)Nx * DX * 2);
  half_t* ym = (half_t*)alloc((size_t)Ly * DY * 2);
  half_t* Wqx = (half_t*)alloc((size_t)3 * DX * DX * 2);
  half_t* Wqy = (half_t*)alloc((size_t)3 * DX * DY * 2);
  half_t* Wpx = (half_t*)alloc((size_t)DX * DX * 2);
  half_t* Wpy = (half_t*)alloc((size_t)DY * DX * 2);
  float* qkvx = (float*)alloc((size_t)Nx * 3 * DX * 4);
  float* qkvy = (float*)alloc((size_t)Ly * 3 * DX * 4);
  half_t* qh = (half_t*)alloc((size_t)AP * DX * 2);
  half_t* kh = (half_t*)alloc((size_t)AP * DX * 2);
  half_t* vT = (half_t*)alloc((size_t)AP * DX * 2);
  half_t* att = (half_t*)alloc((size_t)OUT_ROWS * DX * 2);

  // 1) modulated rmsnorm -> f16
  rmsnorm_scale_cvt<<<Nx, 256, 0, stream>>>(x, scale_x, xm, DX);
  rmsnorm_scale_cvt<<<Ly, 256, 0, stream>>>(y, scale_y, ym, DY);

  // 2) weights -> f16
  cvt_f32_f16<<<4096, 256, 0, stream>>>(W_qkv_x, Wqx, (size_t)3 * DX * DX);
  cvt_f32_f16<<<4096, 256, 0, stream>>>(W_qkv_y, Wqy, (size_t)3 * DX * DY);
  cvt_f32_f16<<<4096, 256, 0, stream>>>(W_proj_x, Wpx, (size_t)DX * DX);
  cvt_f32_f16<<<4096, 256, 0, stream>>>(W_proj_y, Wpy, (size_t)DY * DX);

  // 3) QKV GEMMs (WMMA + TDM staging): [M,K] x [Nt,K]^T
  gemm_f16_wmma<<<dim3(3 * DX / 128, Nx / 128), 256, 0, stream>>>(
      xm, Wqx, b_qkv_x, qkvx, Nx, 3 * DX, DX);
  gemm_f16_wmma<<<dim3(3 * DX / 128, Ly / 128), 256, 0, stream>>>(
      ym, Wqy, b_qkv_y, qkvy, Ly, 3 * DX, DY);

  // 4) q/k norm + RoPE + gather + pad; V transposed per head
  qkv_prep<<<dim3(AP, H), 128, 0, stream>>>(qkvx, qkvy, q_norm_x, k_norm_x,
                                            q_norm_y, k_norm_y, rope_cos,
                                            rope_sin, vti, qh, kh, vT);

  // 5) zero the projection-input rows beyond TOTAL (reference scatters zeros)
  {
    const size_t ntail = (size_t)(OUT_ROWS - TOTAL) * DX;
    fill_zero_f16<<<(unsigned)((ntail + 255) / 256), 256, 0, stream>>>(
        att + (size_t)TOTAL * DX, ntail);
  }

  // 6) flash attention (WMMA): grid = (q tiles, heads), 1 wave each
  flash_attn<<<dim3(AP / 16, H), 32, 0, stream>>>(qh, kh, vT, att);

  // 7) output projections (WMMA) straight into d_out
  gemm_f16_wmma<<<dim3(DX / 128, Nx / 128), 256, 0, stream>>>(
      att, Wpx, b_proj_x, out, Nx, DX, DX);
  gemm_f16_wmma<<<dim3(DY / 128, Ly / 128), 256, 0, stream>>>(
      att + (size_t)Nx * DX, Wpy, b_proj_y, out + (size_t)Nx * DX, Ly, DY, DX);
}
